// DeformableCrossFrameAttention_71957882077582
// MI455X (gfx1250) — matrix operand order repaired and checked by reading, not compiled, not run on hardware
//
#include <hip/hip_runtime.h>
#include <hip/hip_bf16.h>

// ---------------------------------------------------------------------------
// Types for CDNA5 WMMA
// ---------------------------------------------------------------------------
typedef __attribute__((ext_vector_type(16))) __bf16 v16bf;
typedef __attribute__((ext_vector_type(8)))  float  v8f;

union FragBF {
    uint4 u[2];
    v16bf v;
};

__device__ __forceinline__ unsigned short f32_to_bf16(float f) {
    unsigned int u = __float_as_uint(f);
    unsigned int r = u + 0x7FFFu + ((u >> 16) & 1u);   // round-to-nearest-even
    return (unsigned short)(r >> 16);
}

// Direct global->LDS async copy (128-bit per lane), tracked by ASYNCcnt.
// vdst = 32-bit LDS byte address, vaddr = 64-bit global address.
__device__ __forceinline__ void async_copy_b128(unsigned int lds_addr, const void* gaddr) {
    asm volatile("global_load_async_to_lds_b128 %0, %1, off"
                 :: "v"(lds_addr), "v"(gaddr)
                 : "memory");
}

__device__ __forceinline__ void wait_async0() {
    asm volatile("s_wait_asynccnt 0" ::: "memory");
}

// ---------------------------------------------------------------------------
// Conversion kernels
// ---------------------------------------------------------------------------
__global__ __launch_bounds__(256)
void cvt_f32_to_bf16_kernel(const float* __restrict__ in,
                            unsigned short* __restrict__ out, int count) {
    int i = blockIdx.x * 256 + threadIdx.x;
    if (i < count) out[i] = f32_to_bf16(in[i]);
}

// W is (K rows x Ncol cols) row-major f32; produce Wt (Ncol x K) row-major bf16
__global__ __launch_bounds__(256)
void cvt_transpose_bf16_kernel(const float* __restrict__ W,
                               unsigned short* __restrict__ Wt,
                               int K, int Ncol) {
    int i = blockIdx.x * 256 + threadIdx.x;
    int total = K * Ncol;
    if (i >= total) return;
    int n = i / K;
    int k = i - n * K;
    Wt[(size_t)n * K + k] = f32_to_bf16(W[(size_t)k * Ncol + n]);
}

// ---------------------------------------------------------------------------
// bf16 GEMM:  C(MxN,f32) = A(MxK,bf16 row-major) * Bt(NxK,bf16 row-major)^T + bias
// Block tile 128x64, 8 waves (wave32) per block, each wave -> 32x32 (2x2 WMMA).
// Double-buffered LDS filled by GLOBAL_LOAD_ASYNC_TO_LDS_B128 (ASYNCcnt),
// one barrier per K-step, DMA overlapped with v_wmma.
// ---------------------------------------------------------------------------
#define TILE_M 128
#define TILE_N 64
#define TILE_K 32
#define LDSW   40   // element stride (32 data + 8 pad) -> 80B rows, conflict-friendly

__global__ __launch_bounds__(256)
void gemm_bf16_wmma_kernel(const unsigned short* __restrict__ A,
                           const unsigned short* __restrict__ Bt,
                           const float* __restrict__ bias,
                           float* __restrict__ C,
                           int M, int N, int K) {
    __shared__ __align__(16) unsigned short As[2][TILE_M * LDSW];
    __shared__ __align__(16) unsigned short Bs[2][TILE_N * LDSW];

    const int tid  = threadIdx.x;
    const int m0   = blockIdx.y * TILE_M;
    const int n0   = blockIdx.x * TILE_N;
    const int wave = tid >> 5;
    const int lane = tid & 31;
    const int wy   = wave >> 1;          // 0..3 : 32-row band
    const int wx   = wave & 1;           // 0..1 : 32-col band
    const int lr   = lane & 15;          // row/col within 16x16 tile
    const int lhi  = lane >> 4;          // 0 or 1 : K-half selector

    // per-thread staging coordinates (fixed across K-steps)
    const int ar0 = (tid * 2) >> 2;            // A rows for this thread's 2 loads
    const int ac0 = ((tid * 2) & 3) * 8;
    const int ar1 = (tid * 2 + 1) >> 2;
    const int ac1 = ((tid * 2 + 1) & 3) * 8;
    const int br  = tid >> 2;                  // B row
    const int bc  = (tid & 3) * 8;

    auto stage = [&](int buf, int k0) {
        async_copy_b128((unsigned int)(size_t)&As[buf][ar0 * LDSW + ac0],
                        A + (size_t)(m0 + ar0) * K + k0 + ac0);
        async_copy_b128((unsigned int)(size_t)&As[buf][ar1 * LDSW + ac1],
                        A + (size_t)(m0 + ar1) * K + k0 + ac1);
        async_copy_b128((unsigned int)(size_t)&Bs[buf][br * LDSW + bc],
                        Bt + (size_t)(n0 + br) * K + k0 + bc);
    };

    v8f acc[2][2];
#pragma unroll
    for (int ty = 0; ty < 2; ++ty)
#pragma unroll
        for (int tx = 0; tx < 2; ++tx)
            acc[ty][tx] = v8f{0.f, 0.f, 0.f, 0.f, 0.f, 0.f, 0.f, 0.f};

    stage(0, 0);                 // prologue: fill buffer 0
    int cur = 0;
    for (int k0 = 0; k0 < K; k0 += TILE_K) {
        wait_async0();           // own async copies into buf `cur` complete
        __syncthreads();         // everyone's copies done; prev buf fully consumed

        if (k0 + TILE_K < K)     // kick off next tile into the other buffer
            stage(cur ^ 1, k0 + TILE_K);

        // A fragment (16x32 bf16): lanes 0-15 hold K{0..7,16..23}, lanes 16-31 K{8..15,24..31}
        FragBF a[2], b[2];
#pragma unroll
        for (int ty = 0; ty < 2; ++ty) {
            int row = 32 * wy + 16 * ty + lr;
            int e0  = lhi * 8;
            a[ty].u[0] = *(const uint4*)(&As[cur][row * LDSW + e0]);
            a[ty].u[1] = *(const uint4*)(&As[cur][row * LDSW + e0 + 16]);
        }
        // B fragment (32x16 bf16): lanes 0-15 K{0..7,16..23} of col, lanes 16-31 K{8..15,24..31}
#pragma unroll
        for (int tx = 0; tx < 2; ++tx) {
            int row = 32 * wx + 16 * tx + lr;   // row of Bt == output column
            int e0  = lhi * 8;
            b[tx].u[0] = *(const uint4*)(&Bs[cur][row * LDSW + e0]);
            b[tx].u[1] = *(const uint4*)(&Bs[cur][row * LDSW + e0 + 16]);
        }

#pragma unroll
        for (int ty = 0; ty < 2; ++ty)
#pragma unroll
            for (int tx = 0; tx < 2; ++tx)
                acc[ty][tx] = __builtin_amdgcn_wmma_f32_16x16x32_bf16(
                    false, a[ty].v, false, b[tx].v,
                    (short)0, acc[ty][tx], false, false);

        cur ^= 1;
    }

    // epilogue: C/D layout -> reg r, lane l : M = r + 8*(l>=16), N = l%16
#pragma unroll
    for (int ty = 0; ty < 2; ++ty)
#pragma unroll
        for (int tx = 0; tx < 2; ++tx) {
            int col = n0 + 32 * wx + 16 * tx + lr;
            float bv = bias[col];
#pragma unroll
            for (int r = 0; r < 8; ++r) {
                int row = m0 + 32 * wy + 16 * ty + r + 8 * lhi;
                C[(size_t)row * N + col] = acc[ty][tx][r] + bv;
            }
        }
}

// ---------------------------------------------------------------------------
// Deformable sampling + softmax attention.
// One wave32 per (b, n, head); each lane owns channels d and d+32 (hd=64).
// ---------------------------------------------------------------------------
#define NHEADS 8
#define MAXP   8

__device__ __forceinline__ float bgather(const float* __restrict__ imgd,
                                         int yi, int xi, int H, int W) {
    bool valid = (yi >= 0) & (yi < H) & (xi >= 0) & (xi < W);
    int yc = min(max(yi, 0), H - 1);
    int xc = min(max(xi, 0), W - 1);
    float v = imgd[yc * W + xc];
    return valid ? v : 0.0f;
}

__global__ __launch_bounds__(256)
void attn_sample_kernel(const float* __restrict__ Qf,     // (B*N, C)
                        const float* __restrict__ Offs,   // (B*N, 128)
                        const float* __restrict__ ref,    // (B, N, C) == (B, C, H, W)
                        const int* __restrict__ np_ptr,
                        unsigned short* __restrict__ out_bf, // (B*N, C) bf16
                        int B, int N, int C, int H, int W) {
    const int np   = np_ptr[0];
    const int tid  = threadIdx.x;
    const int lane = tid & 31;
    const int gw   = blockIdx.x * 8 + (tid >> 5);
    const int total = B * N * NHEADS;
    if (gw >= total) return;

    const int NHn = N * NHEADS;
    const int b   = gw / NHn;
    const int rem = gw - b * NHn;
    const int n   = rem / NHEADS;
    const int h   = rem - n * NHEADS;
    const int hd  = C / NHEADS;          // 64
    const int HW  = H * W;

    const int d0 = lane;
    const int d1 = lane + 32;

    const float* qrow = Qf + (size_t)(b * N + n) * C + h * hd;
    const float q0 = qrow[d0];
    const float q1 = qrow[d1];

    const float* offrow = Offs + (size_t)(b * N + n) * (NHEADS * MAXP * 2) + h * (MAXP * 2);
    const float* img = ref + (size_t)b * N * C + (size_t)(h * hd) * HW;
    const float* i0  = img + (size_t)d0 * HW;
    const float* i1  = img + (size_t)d1 * HW;

    const int yi_n = n / W;
    const int xi_n = n - yi_n * W;
    const float scale = rsqrtf((float)hd);

    float samp0[MAXP], samp1[MAXP], score[MAXP];

#pragma unroll
    for (int p = 0; p < MAXP; ++p) {
        float s0 = 0.f, s1 = 0.f;
        if (p < np) {
            float off0 = offrow[2 * p];
            float off1 = offrow[2 * p + 1];
            float loc0 = (float)yi_n + off0;
            float loc1 = (float)xi_n + off1;
            // reference's normalize round-trip (keeps its y/x swap semantics)
            float g0 = loc0 / (float)(H - 1) * 2.f - 1.f;
            float g1 = loc1 / (float)(W - 1) * 2.f - 1.f;
            float xp = (g0 + 1.f) * 0.5f * (float)(W - 1);
            float yp = (g1 + 1.f) * 0.5f * (float)(H - 1);
            float x0f = floorf(xp), y0f = floorf(yp);
            int   x0  = (int)x0f,   y0  = (int)y0f;
            float wx1 = xp - x0f, wx0 = 1.f - wx1;
            float wy1 = yp - y0f, wy0 = 1.f - wy1;
            float w00 = wy0 * wx0, w01 = wy0 * wx1;
            float w10 = wy1 * wx0, w11 = wy1 * wx1;
            s0 = bgather(i0, y0, x0, H, W)         * w00
               + bgather(i0, y0, x0 + 1, H, W)     * w01
               + bgather(i0, y0 + 1, x0, H, W)     * w10
               + bgather(i0, y0 + 1, x0 + 1, H, W) * w11;
            s1 = bgather(i1, y0, x0, H, W)         * w00
               + bgather(i1, y0, x0 + 1, H, W)     * w01
               + bgather(i1, y0 + 1, x0, H, W)     * w10
               + bgather(i1, y0 + 1, x0 + 1, H, W) * w11;
        }
        samp0[p] = s0;
        samp1[p] = s1;
        float part = q0 * s0 + q1 * s1;
#pragma unroll
        for (int o = 16; o > 0; o >>= 1)
            part += __shfl_xor(part, o, 32);
        score[p] = (p < np) ? part * scale : -1e30f;
    }

    // softmax over points
    float m = score[0];
#pragma unroll
    for (int p = 1; p < MAXP; ++p) m = fmaxf(m, score[p]);
    float wsum = 0.f;
    float wgt[MAXP];
#pragma unroll
    for (int p = 0; p < MAXP; ++p) {
        float e = __expf(score[p] - m);
        wgt[p] = e;
        wsum += e;
    }
    float inv = 1.f / wsum;
    float o0 = 0.f, o1 = 0.f;
#pragma unroll
    for (int p = 0; p < MAXP; ++p) {
        float a = wgt[p] * inv;
        o0 += a * samp0[p];
        o1 += a * samp1[p];
    }

    unsigned short* orow = out_bf + (size_t)(b * N + n) * C + h * hd;
    orow[d0] = f32_to_bf16(o0);
    orow[d1] = f32_to_bf16(o1);
}

// ---------------------------------------------------------------------------
// Host-side orchestration
// ---------------------------------------------------------------------------
extern "C" void kernel_launch(void* const* d_in, const int* in_sizes, int n_in,
                              void* d_out, int out_size, void* d_ws, size_t ws_size,
                              hipStream_t stream) {
    const float* query = (const float*)d_in[0];
    const float* ref   = (const float*)d_in[1];
    const float* q_w   = (const float*)d_in[2];
    const float* q_b   = (const float*)d_in[3];
    // d_in[4], d_in[5] (kv_w, kv_b) are unused by the reference
    const float* off_w = (const float*)d_in[6];
    const float* off_b = (const float*)d_in[7];
    const float* out_w = (const float*)d_in[8];
    const float* out_b = (const float*)d_in[9];
    const int*   np    = (const int*)d_in[12];   // n_points (device scalar)

    const int C    = in_sizes[3];                // 512
    const int H    = 64, W = 64;                 // per setup_inputs
    const int N    = H * W;                      // 4096
    const int B    = in_sizes[0] / (N * C);      // 4
    const int OFFC = NHEADS * MAXP * 2;          // 128
    const int M    = B * N;                      // 16384

    // workspace carve-out (256B aligned slabs)
    char* ws = (char*)d_ws;
    size_t off = 0;
    auto carve = [&](size_t bytes) -> void* {
        off = (off + 255) & ~(size_t)255;
        void* p = ws + off;
        off += bytes;
        return p;
    };
    unsigned short* qbf   = (unsigned short*)carve((size_t)M * C * 2);
    unsigned short* qwT   = (unsigned short*)carve((size_t)C * C * 2);
    unsigned short* offwT = (unsigned short*)carve((size_t)OFFC * C * 2);
    unsigned short* outwT = (unsigned short*)carve((size_t)C * C * 2);
    float*          Qf    = (float*)carve((size_t)M * C * 4);
    float*          Offs  = (float*)carve((size_t)M * OFFC * 4);
    unsigned short* abf   = (unsigned short*)carve((size_t)M * C * 2);
    (void)ws_size;

    // 1) precision conversion
    {
        int cnt = M * C;
        cvt_f32_to_bf16_kernel<<<(cnt + 255) / 256, 256, 0, stream>>>(query, qbf, cnt);
    }
    cvt_transpose_bf16_kernel<<<(C * C + 255) / 256, 256, 0, stream>>>(q_w,   qwT,   C, C);
    cvt_transpose_bf16_kernel<<<(C * OFFC + 255) / 256, 256, 0, stream>>>(off_w, offwT, C, OFFC);
    cvt_transpose_bf16_kernel<<<(C * C + 255) / 256, 256, 0, stream>>>(out_w, outwT, C, C);

    // 2) q = query @ q_w + q_b            (16384 x 512 x 512)
    gemm_bf16_wmma_kernel<<<dim3(C / TILE_N, M / TILE_M), 256, 0, stream>>>(
        qbf, qwT, q_b, Qf, M, C, C);

    // 3) offsets = query @ off_w + off_b  (16384 x 128 x 512)
    gemm_bf16_wmma_kernel<<<dim3(OFFC / TILE_N, M / TILE_M), 256, 0, stream>>>(
        qbf, offwT, off_b, Offs, M, OFFC, C);

    // 4) deformable sampling + softmax attention (one wave32 per (b,n,head))
    {
        int waves = M * NHEADS;              // 131072
        attn_sample_kernel<<<waves / 8, 256, 0, stream>>>(
            Qf, Offs, ref, np, abf, B, N, C, H, W);
    }

    // 5) out = attn_out @ out_w + out_b   (16384 x 512 x 512) -> d_out (f32)
    gemm_bf16_wmma_kernel<<<dim3(C / TILE_N, M / TILE_M), 256, 0, stream>>>(
        abf, outwT, out_b, (float*)d_out, M, C, C);
}
